// LocationSensitiveAttention_62843961475842
// MI455X (gfx1250) — matrix operand order repaired and checked by reading, not compiled, not run on hardware
//
#include <hip/hip_runtime.h>
#include <hip/hip_bf16.h>
#include <math.h>

typedef __attribute__((ext_vector_type(16))) __bf16          v16bf;
typedef __attribute__((ext_vector_type(8)))  float           v8f;
typedef __attribute__((ext_vector_type(8)))  unsigned short  v8u16;
typedef __attribute__((ext_vector_type(16))) unsigned short  v16u16;

#define B_SZ 64
#define T_SZ 2048
#define E_SZ 768
#define A_SZ 128
#define L_SZ 32
#define K_SZ 31
#define TROWS 32          // T-rows per energies block (2 M-tiles per wave)

__device__ __forceinline__ unsigned short f2bf(float f) {
    union { float f; unsigned int u; } c; c.f = f;
    unsigned int u = c.u;
    return (unsigned short)((u + 0x7FFFu + ((u >> 16) & 1u)) >> 16);
}

// Build a 16-element bf16 A-fragment for V_WMMA_F32_16X16X32_BF16.
// Per ISA 16-bit A layout: a lane's 16 values are two contiguous K-chunks
// [k0, k0+8) and [k0+16, k0+24) of its M-row (k0 = ktile*32 + half*8).
__device__ __forceinline__ v16bf make_afrag(const unsigned short* row, int k0) {
    v8u16 lo = *(const v8u16*)(row + k0);
    v8u16 hi = *(const v8u16*)(row + k0 + 16);
    v16u16 c;
#pragma unroll
    for (int i = 0; i < 8; ++i) { c[i] = lo[i]; c[8 + i] = hi[i]; }
    return __builtin_bit_cast(v16bf, c);
}

// ---------------------------------------------------------------------------
// Pre-swizzle a [ktiles*32, 128] f32 weight into bf16 B-fragments:
// dst[((kt*8 + nt)*32 + lane)*16 + j] <= W[(kt*32 + (lane&16?8:0) + (j&7) + (j&8?16:0))*128 + nt*16 + (lane&15)]
// so each lane reads its fragment as one contiguous 32B chunk at WMMA time.
// ---------------------------------------------------------------------------
__global__ void pack_frags(const float* __restrict__ W, unsigned short* __restrict__ dst,
                           int ktiles) {
    int d = blockIdx.x * 256 + threadIdx.x;
    int total = ktiles * 8 * 32 * 16;
    if (d >= total) return;
    int j     = d & 15;
    int lane  = (d >> 4) & 31;
    int ntile = (d >> 9) & 7;
    int ktile = d >> 12;
    int k = ktile * 32 + ((lane & 16) ? 8 : 0) + (j & 7) + ((j & 8) ? 16 : 0);
    int n = ntile * 16 + (lane & 15);
    dst[d] = f2bf(W[k * A_SZ + n]);
}

// ---------------------------------------------------------------------------
// Ws = decoder_state @ W_w + W_b   ([64,1024] x [1024,128] -> [64,128])
// One wave per 16x16 tile, K looped in steps of 32 (bf16 WMMA).
// ---------------------------------------------------------------------------
__global__ __launch_bounds__(32) void ws_kernel(const float* __restrict__ ds,
                                                const float* __restrict__ Ww,
                                                const float* __restrict__ Wb,
                                                float* __restrict__ Ws) {
    int blk  = blockIdx.x;           // 0..31 : (mt 0..3) x (nt 0..7)
    int mt   = blk >> 3, nt = blk & 7;
    int lane = threadIdx.x;
    int mrow = lane & 15, half = lane >> 4;
    const float* arow = ds + (size_t)(mt * 16 + mrow) * 1024;
    int col = nt * 16 + (lane & 15);
    v8f acc = {};
    for (int kt = 0; kt < 32; ++kt) {
        int k0 = kt * 32 + half * 8;
        v16u16 ar;
#pragma unroll
        for (int j = 0; j < 8; ++j) {
            ar[j]     = f2bf(arow[k0 + j]);
            ar[8 + j] = f2bf(arow[k0 + 16 + j]);
        }
        v16bf a = __builtin_bit_cast(v16bf, ar);
        int kb = kt * 32 + half * 8;
        v16u16 br;
#pragma unroll
        for (int j = 0; j < 16; ++j) br[j] = f2bf(Ww[(size_t)(kb + (j & 7) + ((j & 8) ? 16 : 0)) * A_SZ + col]);
        v16bf bm = __builtin_bit_cast(v16bf, br);
        acc = __builtin_amdgcn_wmma_f32_16x16x32_bf16(false, a, false, bm,
                                                      (short)0, acc, false, false);
    }
    float bias = Wb[col];
#pragma unroll
    for (int r = 0; r < 8; ++r) {
        int row = mt * 16 + half * 8 + r;        // C/D layout: M = r + half*8
        Ws[row * A_SZ + col] = acc[r] + bias;
    }
}

// ---------------------------------------------------------------------------
// energies[b,t] = sum_a tanh(Ws[b,a] + (enc@V_w)[b,t,a] + (loc@U_w)[b,t,a]) * v_w[a]
// One 256-thread block (8 waves) per (b, 32-row T tile). Each wave owns one
// 16-col N tile and TWO 16-row M tiles, so every B fragment loaded from L2
// feeds two WMMAs (halves B-fragment L2 traffic vs 16-row blocks).
// ---------------------------------------------------------------------------
__global__ __launch_bounds__(256) void energies_kernel(
    const float* __restrict__ enc, const float* __restrict__ cum,
    const float* __restrict__ Fw, const float* __restrict__ Ws,
    const float* __restrict__ vw,
    const unsigned short* __restrict__ vfrag, const unsigned short* __restrict__ ufrag,
    float* __restrict__ energies) {
    __shared__ unsigned short enc_s[TROWS][776];   // 768 + 8 pad -> conflict-free
    __shared__ unsigned short loc_s[TROWS][40];    // 32 + 8 pad
    __shared__ float e_lds[TROWS];

    const int tid = threadIdx.x;
    const int b   = blockIdx.y;
    const int tt  = blockIdx.x;                    // 32-row tile index

    // Stage encoding tile [32 x 768] f32 -> bf16 LDS (vectorized 16B loads).
    const float4* src = (const float4*)(enc + (size_t)(b * T_SZ + tt * TROWS) * E_SZ);
#pragma unroll
    for (int i = 0; i < 24; ++i) {
        int flat = tid + i * 256;          // 0..6143 over 32 rows x 192 float4
        int row  = flat / 192;
        int c4   = flat % 192;
        float4 v = src[row * 192 + c4];
        int c = c4 * 4;
        enc_s[row][c]     = f2bf(v.x);
        enc_s[row][c + 1] = f2bf(v.y);
        enc_s[row][c + 2] = f2bf(v.z);
        enc_s[row][c + 3] = f2bf(v.w);
    }
    // Location feature: loc[tl, l] = sum_k F_w[l,k] * cum[b, t + k - 15]
    for (int idx = tid; idx < TROWS * L_SZ; idx += 256) {
        int l = idx & 31, tl = idx >> 5;
        int t = tt * TROWS + tl;
        float s = 0.f;
#pragma unroll
        for (int k = 0; k < K_SZ; ++k) {
            int ti = t + k - 15;
            float c = (ti >= 0 && ti < T_SZ) ? cum[b * T_SZ + ti] : 0.f;
            s = fmaf(Fw[l * K_SZ + k], c, s);
        }
        loc_s[tl][l] = f2bf(s);
    }
    if (tid < TROWS) e_lds[tid] = 0.f;
    __syncthreads();

    const int wave = tid >> 5;               // = N-tile (a-columns wave*16..+15)
    const int lane = tid & 31;
    const int mrow = lane & 15, half = lane >> 4;

    v8f acc0 = {}, acc1 = {};
    for (int kt = 0; kt < 24; ++kt) {        // K = 768 over encoding
        v16u16 braw = *(const v16u16*)(vfrag + (((kt * 8 + wave) * 32 + lane) << 4));
        v16bf bm = __builtin_bit_cast(v16bf, braw);
        int k0 = kt * 32 + half * 8;
        v16bf a0 = make_afrag(&enc_s[mrow][0],      k0);
        v16bf a1 = make_afrag(&enc_s[16 + mrow][0], k0);
        acc0 = __builtin_amdgcn_wmma_f32_16x16x32_bf16(false, a0, false, bm,
                                                       (short)0, acc0, false, false);
        acc1 = __builtin_amdgcn_wmma_f32_16x16x32_bf16(false, a1, false, bm,
                                                       (short)0, acc1, false, false);
    }
    {   // Uf: K = 32 = LOC_DIM in a single WMMA per M tile
        v16u16 braw = *(const v16u16*)(ufrag + ((wave * 32 + lane) << 4));
        v16bf bm = __builtin_bit_cast(v16bf, braw);
        v16bf a0 = make_afrag(&loc_s[mrow][0],      half * 8);
        v16bf a1 = make_afrag(&loc_s[16 + mrow][0], half * 8);
        acc0 = __builtin_amdgcn_wmma_f32_16x16x32_bf16(false, a0, false, bm,
                                                       (short)0, acc0, false, false);
        acc1 = __builtin_amdgcn_wmma_f32_16x16x32_bf16(false, a1, false, bm,
                                                       (short)0, acc1, false, false);
    }

    // Epilogue: + Ws, tanh, * v_w, reduce over a (16 lanes/half + 8 waves).
    int acol  = wave * 16 + (lane & 15);
    float wsv = Ws[b * A_SZ + acol];
    float vv  = vw[acol];
#pragma unroll
    for (int r = 0; r < 8; ++r) {
        float p0 = tanhf(acc0[r] + wsv) * vv;
        float p1 = tanhf(acc1[r] + wsv) * vv;
        p0 += __shfl_xor(p0, 8, 32);
        p0 += __shfl_xor(p0, 4, 32);
        p0 += __shfl_xor(p0, 2, 32);
        p0 += __shfl_xor(p0, 1, 32);
        p1 += __shfl_xor(p1, 8, 32);
        p1 += __shfl_xor(p1, 4, 32);
        p1 += __shfl_xor(p1, 2, 32);
        p1 += __shfl_xor(p1, 1, 32);
        if ((lane & 15) == 0) {
            atomicAdd(&e_lds[half * 8 + r], p0);
            atomicAdd(&e_lds[16 + half * 8 + r], p1);
        }
    }
    __syncthreads();
    if (tid < TROWS) energies[(size_t)b * T_SZ + tt * TROWS + tid] = e_lds[tid];
}

// ---------------------------------------------------------------------------
// Row softmax over T=2048, in place. One block per batch row.
// ---------------------------------------------------------------------------
__global__ __launch_bounds__(256) void softmax_kernel(float* __restrict__ w) {
    __shared__ float red[256];
    int b = blockIdx.x, tid = threadIdx.x;
    float* row = w + (size_t)b * T_SZ;
    float mx = -3.4e38f;
    for (int t = tid; t < T_SZ; t += 256) mx = fmaxf(mx, row[t]);
    red[tid] = mx; __syncthreads();
    for (int s = 128; s > 0; s >>= 1) {
        if (tid < s) red[tid] = fmaxf(red[tid], red[tid + s]);
        __syncthreads();
    }
    mx = red[0]; __syncthreads();
    float sum = 0.f;
    for (int t = tid; t < T_SZ; t += 256) {
        float e = __expf(row[t] - mx);
        row[t] = e; sum += e;
    }
    red[tid] = sum; __syncthreads();
    for (int s = 128; s > 0; s >>= 1) {
        if (tid < s) red[tid] += red[tid + s];
        __syncthreads();
    }
    float inv = 1.f / red[0];
    for (int t = tid; t < T_SZ; t += 256) row[t] *= inv;
}

// ---------------------------------------------------------------------------
// context[b, e] = sum_t w[b,t] * enc[b,t,e]  (second streaming pass over enc)
// ---------------------------------------------------------------------------
__global__ __launch_bounds__(256) void context_kernel(const float* __restrict__ enc,
                                                      const float* __restrict__ w,
                                                      float* __restrict__ out) {
    __shared__ float wsh[128];
    int b = blockIdx.y;
    int e = blockIdx.x * 256 + threadIdx.x;   // 3 blocks/b cover 768 cols
    float acc = 0.f;
    for (int t0 = 0; t0 < T_SZ; t0 += 128) {
        if (threadIdx.x < 128) wsh[threadIdx.x] = w[(size_t)b * T_SZ + t0 + threadIdx.x];
        __syncthreads();
        const float* ep = enc + ((size_t)(b * T_SZ + t0)) * E_SZ + e;
#pragma unroll 8
        for (int k = 0; k < 128; ++k) acc = fmaf(wsh[k], ep[(size_t)k * E_SZ], acc);
        __syncthreads();
    }
    out[(size_t)b * E_SZ + e] = acc;
}

// ---------------------------------------------------------------------------
extern "C" void kernel_launch(void* const* d_in, const int* in_sizes, int n_in,
                              void* d_out, int out_size, void* d_ws, size_t ws_size,
                              hipStream_t stream) {
    const float* decoder_state = (const float*)d_in[0];
    const float* encoding      = (const float*)d_in[1];
    const float* cum           = (const float*)d_in[2];
    const float* W_w           = (const float*)d_in[3];
    const float* W_b           = (const float*)d_in[4];
    const float* V_w           = (const float*)d_in[5];
    const float* U_w           = (const float*)d_in[6];
    const float* F_w           = (const float*)d_in[7];
    const float* v_w           = (const float*)d_in[8];

    float* out     = (float*)d_out;
    float* ctx     = out;                    // [64, 768]
    float* weights = out + B_SZ * E_SZ;      // [64, 2048] (energies -> softmax in place)

    // Workspace layout (bf16 B-fragments + Ws), all 32B-aligned:
    unsigned short* vfrag = (unsigned short*)d_ws;    // 24*8*32*16 = 98304 bf16
    unsigned short* ufrag = vfrag + 98304;            //  1*8*32*16 =  4096 bf16
    float*          Ws    = (float*)(ufrag + 4096);   // 64*128 f32

    pack_frags<<<(98304 + 255) / 256, 256, 0, stream>>>(V_w, vfrag, 24);
    pack_frags<<<(4096  + 255) / 256, 256, 0, stream>>>(U_w, ufrag, 1);
    ws_kernel<<<32, 32, 0, stream>>>(decoder_state, W_w, W_b, Ws);
    energies_kernel<<<dim3(T_SZ / TROWS, B_SZ), 256, 0, stream>>>(
        encoding, cum, F_w, Ws, v_w, vfrag, ufrag, weights);
    softmax_kernel<<<B_SZ, 256, 0, stream>>>(weights);
    context_kernel<<<dim3(3, B_SZ), 256, 0, stream>>>(encoding, weights, ctx);
}
